// MoETransformerLayer_13331578487397
// MI455X (gfx1250) — matrix operand order repaired and checked by reading, not compiled, not run on hardware
//
#include <hip/hip_runtime.h>
#include <cstdint>
#include <cstddef>

// ---------------------------------------------------------------------------
// MoE transformer layer for MI455X (gfx1250, wave32, WMMA).
// Dense algebra -> v_wmma_f32_16x16x32_f16 tiled GEMM, 2x2 register blocking,
// double-buffered LDS staging via GLOBAL_LOAD_ASYNC_TO_LDS_B128 (ASYNCcnt).
// Requires ws_size >= ~512 MB (persistent ~218MB + phase scratch peak ~237MB).
// ---------------------------------------------------------------------------

typedef _Float16 h16;
typedef __attribute__((ext_vector_type(16))) _Float16 v16h;
typedef __attribute__((ext_vector_type(8)))  _Float16 v8h;
typedef __attribute__((ext_vector_type(8)))  float    v8f;

#define CB       32      // batch
#define CL       512     // seq len
#define CD       512     // d_model
#define CT       (CB*CL) // tokens = 16384
#define CH       8       // heads
#define CDK      64      // head dim
#define CDFF     2048
#define CDI      1024    // expand*d_model
#define CDSTATE  16
#define CDTRANK  32
#define CDCONV   4

// ------------------------------ WMMA GEMM ----------------------------------
// C[M,N] = act(alpha * A[M,K] @ op(B) + bias), op(B)=B^T if bT (B stored NxK)
// Block tile 64(M) x 128(N) x 32(K); 256 threads = 8 waves (2x4),
// each wave computes a 32x32 macro-tile = 2x2 WMMA tiles.
#define BM 64
#define BN 128
#define BK 32
#define APITCH 40   // 80 bytes/row -> rows stay 16B aligned for 128b LDS ops
#define BPITCH 40

// async 16-byte global -> LDS copy (per lane); tracked by ASYNCcnt
__device__ __forceinline__ void async_ld16(unsigned lds_off, const h16* gptr)
{
    asm volatile("global_load_async_to_lds_b128 %0, %1, off"
                 :: "v"(lds_off), "v"(gptr) : "memory");
}
__device__ __forceinline__ void wait_async0()
{
    asm volatile("s_wait_asynccnt 0x0" ::: "memory");
}

__global__ __launch_bounds__(256) void gemm_wmma_f16(
    const h16* __restrict__ A, int lda, long long sA,
    const h16* __restrict__ B, int ldb, long long sB, int bT,
    float* __restrict__ Cf, h16* __restrict__ Ch, int ldc, long long sC,
    int M, int N, int K,
    const float* __restrict__ bias, float alpha, int act)
{
    __shared__ h16 As[2][BM * APITCH];
    __shared__ h16 Bs[2][BN * BPITCH];   // always stored [n][k]

    const int tid  = threadIdx.x;
    const int lane = tid & 31;
    const int wave = tid >> 5;
    const int wm   = wave >> 2;          // 0..1  -> 32-row slab
    const int wn   = wave & 3;           // 0..3  -> 32-col slab
    const int m0   = blockIdx.y * BM;
    const int n0   = blockIdx.x * BN;
    const int z    = blockIdx.z;

    const h16* Ab = A + (long long)z * sA;
    const h16* Bb = B + (long long)z * sB;

    const int ml = lane & 15;            // row (A frag) / col (B frag) in tile
    const int g  = lane >> 4;            // lane group 0/1

    v8f acc00 = {}, acc01 = {}, acc10 = {}, acc11 = {};

    // per-thread staging coordinates (one/two 16B chunks per thread)
    const int srow = tid >> 2;           // 0..63
    const int scol = (tid & 3) * 8;      // 0,8,16,24

    auto compute_stage = [&](const h16* Asb, const h16* Bsb) {
        v16h af[2], bf[2];
#pragma unroll
        for (int sm = 0; sm < 2; ++sm) {
            const h16* ap = Asb + (wm * 32 + sm * 16 + ml) * APITCH;
            v8h alo = *(const v8h*)(ap + 8 * g);
            v8h ahi = *(const v8h*)(ap + 16 + 8 * g);
#pragma unroll
            for (int j = 0; j < 8; ++j) { af[sm][j] = alo[j]; af[sm][j + 8] = ahi[j]; }
        }
#pragma unroll
        for (int sn = 0; sn < 2; ++sn) {
            const h16* bp = Bsb + (wn * 32 + sn * 16 + ml) * BPITCH;
            v8h blo = *(const v8h*)(bp + 8 * g);
            v8h bhi = *(const v8h*)(bp + 16 + 8 * g);
#pragma unroll
            for (int j = 0; j < 8; ++j) { bf[sn][j] = blo[j]; bf[sn][j + 8] = bhi[j]; }
        }
        acc00 = __builtin_amdgcn_wmma_f32_16x16x32_f16(false, af[0], false, bf[0], (short)0, acc00, false, false);
        acc01 = __builtin_amdgcn_wmma_f32_16x16x32_f16(false, af[0], false, bf[1], (short)0, acc01, false, false);
        acc10 = __builtin_amdgcn_wmma_f32_16x16x32_f16(false, af[1], false, bf[0], (short)0, acc10, false, false);
        acc11 = __builtin_amdgcn_wmma_f32_16x16x32_f16(false, af[1], false, bf[1], (short)0, acc11, false, false);
    };

    const bool pipelined = bT && (M % BM == 0) && (N % BN == 0) && (K % BK == 0);

    if (pipelined) {
        // -------- double-buffered async pipeline (all tiles in-bounds) -----
        auto stage = [&](int t, int bufi) {
            const int k0 = t * BK;
            async_ld16((unsigned)(uintptr_t)&As[bufi][srow * APITCH + scol],
                       Ab + (long long)(m0 + srow) * lda + (k0 + scol));
            async_ld16((unsigned)(uintptr_t)&Bs[bufi][srow * BPITCH + scol],
                       Bb + (long long)(n0 + srow) * ldb + (k0 + scol));
            async_ld16((unsigned)(uintptr_t)&Bs[bufi][(64 + srow) * BPITCH + scol],
                       Bb + (long long)(n0 + 64 + srow) * ldb + (k0 + scol));
        };
        const int nt = K / BK;
        stage(0, 0);
        wait_async0();
        __syncthreads();
        for (int t = 0; t < nt; ++t) {
            const int cur = t & 1;
            if (t + 1 < nt) stage(t + 1, cur ^ 1);   // overlap with compute
            compute_stage(&As[cur][0], &Bs[cur][0]);
            wait_async0();
            __syncthreads();
        }
    } else {
        // -------- generic path with bounds checks (N=64 GEMMs, tails) ------
        for (int k0 = 0; k0 < K; k0 += BK) {
            const bool fullK = (k0 + BK) <= K;
            if (fullK && (m0 + BM) <= M) {
                *(v8h*)(&As[0][srow * APITCH + scol]) =
                    *(const v8h*)(Ab + (long long)(m0 + srow) * lda + (k0 + scol));
            } else {
                for (int idx = tid; idx < BM * BK; idx += 256) {
                    int r = idx >> 5, c = idx & 31;
                    int gm = m0 + r, gk = k0 + c;
                    h16 v = (h16)0.f;
                    if (gm < M && gk < K) v = Ab[(long long)gm * lda + gk];
                    As[0][r * APITCH + c] = v;
                }
            }
            if (bT) {
                if (fullK && (n0 + BN) <= N) {
#pragma unroll
                    for (int q = 0; q < 2; ++q) {
                        int row = q * 64 + srow;
                        *(v8h*)(&Bs[0][row * BPITCH + scol]) =
                            *(const v8h*)(Bb + (long long)(n0 + row) * ldb + (k0 + scol));
                    }
                } else {
                    for (int idx = tid; idx < BN * BK; idx += 256) {
                        int r = idx >> 5, c = idx & 31;
                        int gn = n0 + r, gk = k0 + c;
                        h16 v = (h16)0.f;
                        if (gn < N && gk < K) v = Bb[(long long)gn * ldb + gk];
                        Bs[0][r * BPITCH + c] = v;
                    }
                }
            } else {
                if (fullK && (n0 + BN) <= N) {
#pragma unroll
                    for (int q = 0; q < 2; ++q) {
                        int idx = tid + q * 256;
                        int kk = idx >> 4;            // 0..31
                        int nn = (idx & 15) * 8;      // 0..120
                        v8h v = *(const v8h*)(Bb + (long long)(k0 + kk) * ldb + (n0 + nn));
#pragma unroll
                        for (int e = 0; e < 8; ++e)
                            Bs[0][(nn + e) * BPITCH + kk] = v[e];
                    }
                } else {
                    for (int idx = tid; idx < BN * BK; idx += 256) {
                        int kk = idx >> 7;            // 0..31
                        int nn = idx & 127;           // 0..127
                        int gn = n0 + nn, gk = k0 + kk;
                        h16 v = (h16)0.f;
                        if (gn < N && gk < K) v = Bb[(long long)gk * ldb + gn];
                        Bs[0][nn * BPITCH + kk] = v;
                    }
                }
            }
            __syncthreads();
            compute_stage(&As[0][0], &Bs[0][0]);
            __syncthreads();
        }
    }

    // ---- epilogue: acc element r -> row 8g+r of its tile, col = ml ----
    v8f* accs[4] = {&acc00, &acc01, &acc10, &acc11};
#pragma unroll
    for (int sm = 0; sm < 2; ++sm) {
#pragma unroll
        for (int sn = 0; sn < 2; ++sn) {
            v8f& acc = *accs[sm * 2 + sn];
            const int gn = n0 + wn * 32 + sn * 16 + ml;
#pragma unroll
            for (int r = 0; r < 8; ++r) {
                int gm = m0 + wm * 32 + sm * 16 + g * 8 + r;
                if (gm < M && gn < N) {
                    float v = acc[r] * alpha;
                    if (bias) v += bias[gn];
                    if (act == 1)      v = fmaxf(v, 0.f);                        // relu
                    else if (act == 2) v = v / (1.f + __expf(-v));               // silu
                    else if (act == 3) v = (v > 20.f) ? v : log1pf(__expf(v));   // softplus
                    long long off = (long long)z * sC + (long long)gm * ldc + gn;
                    if (Cf) Cf[off] = v;
                    if (Ch) Ch[off] = (h16)v;
                }
            }
        }
    }
}

// --------------------------- elementwise kernels ---------------------------
__device__ __forceinline__ float sigf(float x) { return 1.f / (1.f + __expf(-x)); }

__global__ void k_f32_to_f16(const float* __restrict__ in, h16* __restrict__ out, long long n)
{
    long long i = (long long)blockIdx.x * blockDim.x + threadIdx.x;
    long long st = (long long)gridDim.x * blockDim.x;
    for (; i < n; i += st) out[i] = (h16)in[i];
}

__global__ void k_add2(const float* a, const float* b, float* o, int n)
{
    int i = blockIdx.x * blockDim.x + threadIdx.x;
    if (i < n) o[i] = a[i] + b[i];
}

// na_in[t, 0:512]=query ; [512:1024] = values shifted right by 1 within seq
__global__ void k_build_na_in(const float* __restrict__ q, const float* __restrict__ v,
                              h16* __restrict__ out)
{
    long long i = (long long)blockIdx.x * blockDim.x + threadIdx.x;
    long long st = (long long)gridDim.x * blockDim.x;
    for (; i < (long long)CT * 1024; i += st) {
        long long t = i >> 10; int c = (int)(i & 1023);
        int l = (int)(t & (CL - 1));
        float val;
        if (c < CD) val = q[t * CD + c];
        else        val = (l > 0) ? v[(t - 1) * CD + (c - CD)] : 0.f;
        out[i] = (h16)val;
    }
}

// routing: softmax over 4 gate logits, entropy-confidence top-k weights
__global__ void k_routing(const float* __restrict__ q, const float* __restrict__ gW,
                          const float* __restrict__ gb,
                          float* __restrict__ nw, int* __restrict__ ti)
{
    int t = blockIdx.x * blockDim.x + threadIdx.x;
    if (t >= CT) return;
    const float* x = q + (long long)t * CD;
    float lg[4];
    for (int e = 0; e < 4; ++e) {
        float a = gb[e];
        const float* w = gW + e * CD;
        for (int k = 0; k < CD; ++k) a += x[k] * w[k];
        lg[e] = a;
    }
    float mx = fmaxf(fmaxf(lg[0], lg[1]), fmaxf(lg[2], lg[3]));
    float p[4], s = 0.f;
    for (int e = 0; e < 4; ++e) { p[e] = __expf(lg[e] - mx); s += p[e]; }
    float ent = 0.f;
    for (int e = 0; e < 4; ++e) { p[e] /= s; ent -= p[e] * __logf(p[e] + 1e-8f); }
    float conf = 1.f - ent / 1.3862943611198906f;
    int kv = (conf >= 0.8f) ? 1 : ((conf >= 0.6f) ? 2 : ((conf >= 0.4f) ? 3 : 4));
    int idx[4] = {0, 1, 2, 3};
    for (int i = 0; i < 4; ++i)
        for (int j = i + 1; j < 4; ++j)
            if (p[idx[j]] > p[idx[i]]) { int tm = idx[i]; idx[i] = idx[j]; idx[j] = tm; }
    float w4[4], ss = 0.f;
    for (int i = 0; i < 4; ++i) { w4[i] = (i < kv) ? p[idx[i]] : 0.f; ss += w4[i]; }
    for (int i = 0; i < 4; ++i) { nw[t * 4 + i] = w4[i] / (ss + 1e-8f); ti[t * 4 + i] = idx[i]; }
}

// im2col for causal conv (kernel 3, left pad 2): out[t, i*3+tap] = na[t+tap-2, i]
__global__ void k_im2col3(const h16* __restrict__ na, h16* __restrict__ out)
{
    long long i = (long long)blockIdx.x * blockDim.x + threadIdx.x;
    long long st = (long long)gridDim.x * blockDim.x;
    for (; i < (long long)CT * 1536; i += st) {
        long long t = i / 1536; int c = (int)(i % 1536);
        int ii = c / 3, tap = c % 3;
        int l = (int)(t & (CL - 1)), ls = l + tap - 2;
        h16 v = (h16)0.f;
        if (ls >= 0) v = na[(t + tap - 2) * CD + ii];
        out[i] = v;
    }
}

// LSTM scan: 1 block per batch, 512 threads (one hidden channel each);
// Whh rows read with 128-bit loads, h kept in LDS.
__global__ __launch_bounds__(512) void k_lstm_scan(const float* __restrict__ xg,
                                                   const h16* __restrict__ Whh,
                                                   float* __restrict__ out)
{
    __shared__ float h[CD];
    int b = blockIdx.x, j = threadIdx.x;
    h[j] = 0.f; float c = 0.f;
    __syncthreads();
    const float* xgb = xg + (long long)b * CL * (4 * CD);
    float* ob = out + (long long)b * CL * CD;
    const h16* wi = Whh + (long long)(0 * CD + j) * CD;
    const h16* wf = Whh + (long long)(1 * CD + j) * CD;
    const h16* wg = Whh + (long long)(2 * CD + j) * CD;
    const h16* wo = Whh + (long long)(3 * CD + j) * CD;
    for (int t = 0; t < CL; ++t) {
        const float* xt = xgb + (long long)t * (4 * CD);
        float gi = xt[j], gf = xt[CD + j], gg = xt[2 * CD + j], go = xt[3 * CD + j];
        for (int k = 0; k < CD; k += 8) {
            v8h a = *(const v8h*)(wi + k);
            v8h f = *(const v8h*)(wf + k);
            v8h gv = *(const v8h*)(wg + k);
            v8h o = *(const v8h*)(wo + k);
#pragma unroll
            for (int e = 0; e < 8; ++e) {
                float hv = h[k + e];
                gi += hv * (float)a[e]; gf += hv * (float)f[e];
                gg += hv * (float)gv[e]; go += hv * (float)o[e];
            }
        }
        c = sigf(gf) * c + sigf(gi) * tanhf(gg);
        float hn = sigf(go) * tanhf(c);
        __syncthreads();
        h[j] = hn;
        __syncthreads();
        ob[(long long)t * CD + j] = hn;
    }
}

// causal softmax (strict lower triangular, row 0 -> zeros), for one batch
__global__ __launch_bounds__(256) void k_attn_softmax(const float* __restrict__ scores,
                                                      h16* __restrict__ probs)
{
    int i = blockIdx.x, hh = blockIdx.y, tid = threadIdx.x;
    const float* sr = scores + ((long long)hh * CL + i) * CL;
    h16* pr = probs + ((long long)hh * CL + i) * CL;
    if (i == 0) { for (int j = tid; j < CL; j += 256) pr[j] = (h16)0.f; return; }
    __shared__ float red[256];
    int n = i;
    float mx = -3.0e38f;
    for (int j = tid; j < n; j += 256) mx = fmaxf(mx, sr[j]);
    red[tid] = mx; __syncthreads();
    for (int s = 128; s > 0; s >>= 1) { if (tid < s) red[tid] = fmaxf(red[tid], red[tid + s]); __syncthreads(); }
    mx = red[0]; __syncthreads();
    float sum = 0.f;
    for (int j = tid; j < n; j += 256) sum += __expf(sr[j] - mx);
    red[tid] = sum; __syncthreads();
    for (int s = 128; s > 0; s >>= 1) { if (tid < s) red[tid] += red[tid + s]; __syncthreads(); }
    float inv = 1.f / red[0];
    for (int j = tid; j < CL; j += 256)
        pr[j] = (h16)((j < n) ? __expf(sr[j] - mx) * inv : 0.f);
}

// mamba depthwise causal conv (kernel 4, left pad 3) + SiLU -> f16
__global__ void k_mamba_conv(const float* __restrict__ xz, const float* __restrict__ w,
                             const float* __restrict__ bias, h16* __restrict__ xi)
{
    long long i = (long long)blockIdx.x * blockDim.x + threadIdx.x;
    long long st = (long long)gridDim.x * blockDim.x;
    for (; i < (long long)CT * CDI; i += st) {
        long long t = i >> 10; int d = (int)(i & (CDI - 1));
        int l = (int)(t & (CL - 1));
        float acc = bias[d];
        for (int tap = 0; tap < CDCONV; ++tap) {
            int ls = l - 3 + tap;
            if (ls >= 0) acc += xz[(t - 3 + tap) * (2 * CDI) + d] * w[d * CDCONV + tap];
        }
        acc = acc * sigf(acc);
        xi[i] = (h16)acc;
    }
}

// mamba selective scan, fused with D-skip and SiLU(z) gate. 1 block/batch,
// 256 threads each owning 4 channels * 16 states in registers.
__global__ __launch_bounds__(256) void k_mamba_scan(
    const h16* __restrict__ xi, const h16* __restrict__ dt16,
    const h16* __restrict__ dbl, const float* __restrict__ A_log,
    const float* __restrict__ Dp, const float* __restrict__ xz,
    h16* __restrict__ yy)
{
    int b = blockIdx.x, j = threadIdx.x;
    __shared__ float Bt[CDSTATE], Ct[CDSTATE];
    float hst[4][CDSTATE], Ad[4][CDSTATE];
    for (int u = 0; u < 4; ++u)
        for (int s = 0; s < CDSTATE; ++s) {
            hst[u][s] = 0.f;
            Ad[u][s] = -__expf(A_log[(j * 4 + u) * CDSTATE + s]);
        }
    for (int t = 0; t < CL; ++t) {
        long long tok = (long long)b * CL + t;
        if (j < CDSTATE)            Bt[j] = (float)dbl[tok * 64 + CDTRANK + j];
        else if (j < 2 * CDSTATE)   Ct[j - CDSTATE] = (float)dbl[tok * 64 + CDTRANK + CDSTATE + (j - CDSTATE)];
        __syncthreads();
        for (int u = 0; u < 4; ++u) {
            int d = j * 4 + u;
            float dtv = (float)dt16[tok * CDI + d];
            float xv  = (float)xi[tok * CDI + d];
            float dx  = dtv * xv;
            float acc = 0.f;
#pragma unroll
            for (int s = 0; s < CDSTATE; ++s) {
                float hh = __expf(dtv * Ad[u][s]) * hst[u][s] + dx * Bt[s];
                hst[u][s] = hh;
                acc += hh * Ct[s];
            }
            float zg = xz[tok * (2 * CDI) + CDI + d];
            float yv = (acc + xv * Dp[d]) * (zg * sigf(zg));
            yy[tok * CDI + d] = (h16)yv;
        }
        __syncthreads();
    }
}

// moe combine + residual + layernorm (1 block per token)
__global__ __launch_bounds__(256) void k_moe_combine_ln(
    const float* __restrict__ q, const float* __restrict__ nw, const int* __restrict__ ti,
    const float* __restrict__ e0, const float* __restrict__ e1,
    const float* __restrict__ e2, const float* __restrict__ e3,
    const float* __restrict__ g, const float* __restrict__ be,
    float* __restrict__ xo, h16* __restrict__ xo16)
{
    long long t = blockIdx.x;
    int tid = threadIdx.x;
    const float* ex[4] = {e0, e1, e2, e3};
    __shared__ float r1[256], r2[256];
    float vals[2], s = 0.f, s2 = 0.f;
    for (int u = 0; u < 2; ++u) {
        int d = tid + u * 256;
        float m = 0.f;
        for (int e = 0; e < 4; ++e)
            m += nw[t * 4 + e] * ex[ti[t * 4 + e]][t * CD + d];
        float v = q[t * CD + d] + m;
        vals[u] = v; s += v; s2 += v * v;
    }
    r1[tid] = s; r2[tid] = s2; __syncthreads();
    for (int st = 128; st > 0; st >>= 1) {
        if (tid < st) { r1[tid] += r1[tid + st]; r2[tid] += r2[tid + st]; }
        __syncthreads();
    }
    float mean = r1[0] / (float)CD;
    float var  = r2[0] / (float)CD - mean * mean;
    float inv  = rsqrtf(var + 1e-5f);
    for (int u = 0; u < 2; ++u) {
        int d = tid + u * 256;
        float v = (vals[u] - mean) * inv * g[d] + be[d];
        xo[t * CD + d] = v;
        xo16[t * CD + d] = (h16)v;
    }
}

// out = layernorm(x + f) * g + b   (1 block per token)
__global__ __launch_bounds__(256) void k_add_ln(
    const float* __restrict__ x, const float* __restrict__ f,
    const float* __restrict__ g, const float* __restrict__ be,
    float* __restrict__ out)
{
    long long t = blockIdx.x;
    int tid = threadIdx.x;
    __shared__ float r1[256], r2[256];
    float vals[2], s = 0.f, s2 = 0.f;
    for (int u = 0; u < 2; ++u) {
        int d = tid + u * 256;
        float v = x[t * CD + d] + f[t * CD + d];
        vals[u] = v; s += v; s2 += v * v;
    }
    r1[tid] = s; r2[tid] = s2; __syncthreads();
    for (int st = 128; st > 0; st >>= 1) {
        if (tid < st) { r1[tid] += r1[tid + st]; r2[tid] += r2[tid + st]; }
        __syncthreads();
    }
    float mean = r1[0] / (float)CD;
    float var  = r2[0] / (float)CD - mean * mean;
    float inv  = rsqrtf(var + 1e-5f);
    for (int u = 0; u < 2; ++u) {
        int d = tid + u * 256;
        out[t * CD + d] = (vals[u] - mean) * inv * g[d] + be[d];
    }
}

// ------------------------------ host driver --------------------------------
static inline void launch_gemm(hipStream_t s,
    const h16* A, int lda, long long sA,
    const h16* B, int ldb, long long sB, int bT,
    float* Cf, h16* Ch, int ldc, long long sC,
    int M, int N, int K, const float* bias, float alpha, int act, int batch)
{
    dim3 grid((N + BN - 1) / BN, (M + BM - 1) / BM, batch);
    gemm_wmma_f16<<<grid, dim3(256), 0, s>>>(A, lda, sA, B, ldb, sB, bT,
                                             Cf, Ch, ldc, sC, M, N, K, bias, alpha, act);
}

static inline void launch_cvt(hipStream_t s, const float* in, h16* out, long long n)
{
    long long blocks = (n + 255) / 256;
    if (blocks > 32768) blocks = 32768;
    k_f32_to_f16<<<dim3((unsigned)blocks), dim3(256), 0, s>>>(in, out, n);
}

extern "C" void kernel_launch(void* const* d_in, const int* in_sizes, int n_in,
                              void* d_out, int out_size, void* d_ws, size_t ws_size,
                              hipStream_t stream)
{
    (void)in_sizes; (void)n_in; (void)out_size; (void)ws_size;
    const float* query   = (const float*)d_in[0];
    const float* key     = (const float*)d_in[1];
    const float* values  = (const float*)d_in[2];
    const float* gate_W  = (const float*)d_in[3];
    const float* gate_b  = (const float*)d_in[4];
    const float* concat_W= (const float*)d_in[5];
    const float* k_W     = (const float*)d_in[6];
    const float* k_b     = (const float*)d_in[7];
    const float* v_W     = (const float*)d_in[8];
    const float* v_b     = (const float*)d_in[9];
    const float* out_W   = (const float*)d_in[10];
    const float* out_b   = (const float*)d_in[11];
    const float* cnn_W   = (const float*)d_in[12];
    const float* cnn_b   = (const float*)d_in[13];
    const float* lstm_Wih= (const float*)d_in[14];
    const float* lstm_Whh= (const float*)d_in[15];
    const float* lstm_bih= (const float*)d_in[16];
    const float* lstm_bhh= (const float*)d_in[17];
    const float* m_in_W  = (const float*)d_in[18];
    const float* m_conv_W= (const float*)d_in[19];
    const float* m_conv_b= (const float*)d_in[20];
    const float* m_x_W   = (const float*)d_in[21];
    const float* m_dt_W  = (const float*)d_in[22];
    const float* m_dt_b  = (const float*)d_in[23];
    const float* m_A_log = (const float*)d_in[24];
    const float* m_D     = (const float*)d_in[25];
    const float* m_out_W = (const float*)d_in[26];
    const float* ff_W1   = (const float*)d_in[27];
    const float* ff_b1   = (const float*)d_in[28];
    const float* ff_W2   = (const float*)d_in[29];
    const float* ff_b2   = (const float*)d_in[30];
    const float* ln1_g   = (const float*)d_in[31];
    const float* ln1_b   = (const float*)d_in[32];
    const float* ln2_g   = (const float*)d_in[33];
    const float* ln2_b   = (const float*)d_in[34];
    float* out = (float*)d_out;

    char* base = (char*)d_ws;
    size_t cur = 0;
    auto alloc = [&](size_t bytes) -> void* {
        cur = (cur + 255) & ~(size_t)255;
        void* p = base + cur;
        cur += bytes;
        return p;
    };

    // ---- persistent region ----
    h16* cW16   = (h16*)alloc((size_t)CD * 1024 * 2);
    h16* kW16   = (h16*)alloc((size_t)CD * CD * 2);
    h16* vW16   = (h16*)alloc((size_t)CD * CD * 2);
    h16* oW16   = (h16*)alloc((size_t)CD * CD * 2);
    h16* cnnW16 = (h16*)alloc((size_t)CD * 1536 * 2);
    h16* Wih16  = (h16*)alloc((size_t)4 * CD * CD * 2);
    h16* Whh16  = (h16*)alloc((size_t)4 * CD * CD * 2);
    h16* inW16  = (h16*)alloc((size_t)2 * CDI * CD * 2);
    h16* xW16   = (h16*)alloc((size_t)64 * CDI * 2);
    h16* dtW16  = (h16*)alloc((size_t)CDI * CDTRANK * 2);
    h16* moW16  = (h16*)alloc((size_t)CD * CDI * 2);
    h16* f1W16  = (h16*)alloc((size_t)CDFF * CD * 2);
    h16* f2W16  = (h16*)alloc((size_t)CD * CDFF * 2);
    float* lbias = (float*)alloc((size_t)4 * CD * 4);
    float* nw    = (float*)alloc((size_t)CT * 4 * 4);
    int*   ti    = (int*)alloc((size_t)CT * 4 * 4);
    h16* na16    = (h16*)alloc((size_t)CT * CD * 2);
    float* e_cnn  = (float*)alloc((size_t)CT * CD * 4);
    float* e_lstm = (float*)alloc((size_t)CT * CD * 4);
    float* e_attn = (float*)alloc((size_t)CT * CD * 4);
    float* e_mamba= (float*)alloc((size_t)CT * CD * 4);
    float* x32    = (float*)alloc((size_t)CT * CD * 4);
    h16*   x16    = (h16*)alloc((size_t)CT * CD * 2);
    const size_t mark = cur;
    auto reset = [&]() { cur = mark; };

    // ---- weight conversions ----
    launch_cvt(stream, concat_W, cW16,  (long long)CD * 1024);
    launch_cvt(stream, k_W,      kW16,  (long long)CD * CD);
    launch_cvt(stream, v_W,      vW16,  (long long)CD * CD);
    launch_cvt(stream, out_W,    oW16,  (long long)CD * CD);
    launch_cvt(stream, cnn_W,    cnnW16,(long long)CD * 1536);
    launch_cvt(stream, lstm_Wih, Wih16, (long long)4 * CD * CD);
    launch_cvt(stream, lstm_Whh, Whh16, (long long)4 * CD * CD);
    launch_cvt(stream, m_in_W,   inW16, (long long)2 * CDI * CD);
    launch_cvt(stream, m_x_W,    xW16,  (long long)64 * CDI);
    launch_cvt(stream, m_dt_W,   dtW16, (long long)CDI * CDTRANK);
    launch_cvt(stream, m_out_W,  moW16, (long long)CD * CDI);
    launch_cvt(stream, ff_W1,    f1W16, (long long)CDFF * CD);
    launch_cvt(stream, ff_W2,    f2W16, (long long)CD * CDFF);
    k_add2<<<dim3((4 * CD + 255) / 256), dim3(256), 0, stream>>>(lstm_bih, lstm_bhh, lbias, 4 * CD);

    // ---- na = concat(query, hist) @ concat_W^T ----
    {
        h16* na_in16 = (h16*)alloc((size_t)CT * 1024 * 2);
        k_build_na_in<<<dim3(32768), dim3(256), 0, stream>>>(query, values, na_in16);
        launch_gemm(stream, na_in16, 1024, 0, cW16, 1024, 0, 1,
                    nullptr, na16, CD, 0, CT, CD, 1024, nullptr, 1.f, 0, 1);
        reset();
    }

    // ---- routing ----
    k_routing<<<dim3((CT + 127) / 128), dim3(128), 0, stream>>>(query, gate_W, gate_b, nw, ti);

    // ---- CNN expert: im2col + GEMM (K=1536) + bias + relu ----
    {
        h16* col16 = (h16*)alloc((size_t)CT * 1536 * 2);
        k_im2col3<<<dim3(32768), dim3(256), 0, stream>>>(na16, col16);
        launch_gemm(stream, col16, 1536, 0, cnnW16, 1536, 0, 1,
                    e_cnn, nullptr, CD, 0, CT, CD, 1536, cnn_b, 1.f, 1, 1);
        reset();
    }

    // ---- LSTM expert: xg GEMM then sequential scan ----
    {
        float* xg = (float*)alloc((size_t)CT * 4 * CD * 4);
        launch_gemm(stream, na16, CD, 0, Wih16, CD, 0, 1,
                    xg, nullptr, 4 * CD, 0, CT, 4 * CD, CD, lbias, 1.f, 0, 1);
        k_lstm_scan<<<dim3(CB), dim3(512), 0, stream>>>(xg, Whh16, e_lstm);
        reset();
    }

    // ---- Attention expert ----
    {
        h16* q16  = (h16*)alloc((size_t)CT * CD * 2);
        h16* k16  = (h16*)alloc((size_t)CT * CD * 2);
        h16* v16  = (h16*)alloc((size_t)CT * CD * 2);
        h16* qp16 = (h16*)alloc((size_t)CT * CD * 2);
        h16* kp16 = (h16*)alloc((size_t)CT * CD * 2);
        h16* vp16 = (h16*)alloc((size_t)CT * CD * 2);
        float* scores = (float*)alloc((size_t)CH * CL * CL * 4);
        h16* probs16  = (h16*)alloc((size_t)CH * CL * CL * 2);
        h16* ao16     = (h16*)alloc((size_t)CT * CD * 2);
        launch_cvt(stream, query,  q16, (long long)CT * CD);
        launch_cvt(stream, key,    k16, (long long)CT * CD);
        launch_cvt(stream, values, v16, (long long)CT * CD);
        launch_gemm(stream, q16, CD, 0, kW16, CD, 0, 1, nullptr, qp16, CD, 0, CT, CD, CD, k_b, 1.f, 0, 1);
        launch_gemm(stream, k16, CD, 0, kW16, CD, 0, 1, nullptr, kp16, CD, 0, CT, CD, CD, k_b, 1.f, 0, 1);
        launch_gemm(stream, v16, CD, 0, vW16, CD, 0, 1, nullptr, vp16, CD, 0, CT, CD, CD, v_b, 1.f, 0, 1);
        for (int b = 0; b < CB; ++b) {
            const h16* qb = qp16 + (long long)b * CL * CD;
            const h16* kb = kp16 + (long long)b * CL * CD;
            const h16* vb = vp16 + (long long)b * CL * CD;
            // scores[h] = (q_h @ k_h^T) / 8 ; head h = columns h*64..h*64+63
            launch_gemm(stream, qb, CD, CDK, kb, CD, CDK, 1,
                        scores, nullptr, CL, (long long)CL * CL,
                        CL, CL, CDK, nullptr, 0.125f, 0, CH);
            k_attn_softmax<<<dim3(CL, CH), dim3(256), 0, stream>>>(scores, probs16);
            // attn_out[h] = probs @ v_h   (B in KxN form)
            launch_gemm(stream, probs16, CL, (long long)CL * CL, vb, CD, CDK, 0,
                        nullptr, ao16 + (long long)b * CL * CD, CD, CDK,
                        CL, CDK, CL, nullptr, 1.f, 0, CH);
        }
        launch_gemm(stream, ao16, CD, 0, oW16, CD, 0, 1,
                    e_attn, nullptr, CD, 0, CT, CD, CD, out_b, 1.f, 0, 1);
        reset();
    }

    // ---- Mamba expert ----
    {
        float* xz   = (float*)alloc((size_t)CT * 2 * CDI * 4);
        h16*   xi16 = (h16*)alloc((size_t)CT * CDI * 2);
        h16*   dbl16= (h16*)alloc((size_t)CT * 64 * 2);
        h16*   dt16 = (h16*)alloc((size_t)CT * CDI * 2);
        h16*   yy16 = (h16*)alloc((size_t)CT * CDI * 2);
        launch_gemm(stream, na16, CD, 0, inW16, CD, 0, 1,
                    xz, nullptr, 2 * CDI, 0, CT, 2 * CDI, CD, nullptr, 1.f, 0, 1);
        k_mamba_conv<<<dim3(32768), dim3(256), 0, stream>>>(xz, m_conv_W, m_conv_b, xi16);
        launch_gemm(stream, xi16, CDI, 0, xW16, CDI, 0, 1,
                    nullptr, dbl16, 64, 0, CT, 64, CDI, nullptr, 1.f, 0, 1);
        launch_gemm(stream, dbl16, 64, 0, dtW16, CDTRANK, 0, 1,
                    nullptr, dt16, CDI, 0, CT, CDI, CDTRANK, m_dt_b, 1.f, 3, 1);
        k_mamba_scan<<<dim3(CB), dim3(256), 0, stream>>>(xi16, dt16, dbl16, m_A_log, m_D, xz, yy16);
        launch_gemm(stream, yy16, CDI, 0, moW16, CDI, 0, 1,
                    e_mamba, nullptr, CD, 0, CT, CD, CDI, nullptr, 1.f, 0, 1);
        reset();
    }

    // ---- MoE combine + residual + LN1 ----
    k_moe_combine_ln<<<dim3(CT), dim3(256), 0, stream>>>(
        query, nw, ti, e_cnn, e_lstm, e_attn, e_mamba, ln1_g, ln1_b, x32, x16);

    // ---- FFN + residual + LN2 ----
    {
        h16* ff1 = (h16*)alloc((size_t)CT * CDFF * 2);
        float* ff2 = (float*)alloc((size_t)CT * CD * 4);
        launch_gemm(stream, x16, CD, 0, f1W16, CD, 0, 1,
                    nullptr, ff1, CDFF, 0, CT, CDFF, CD, ff_b1, 1.f, 1, 1);
        launch_gemm(stream, ff1, CDFF, 0, f2W16, CDFF, 0, 1,
                    ff2, nullptr, CD, 0, CT, CD, CDFF, ff_b2, 1.f, 0, 1);
        k_add_ln<<<dim3(CT), dim3(256), 0, stream>>>(x32, ff2, ln2_g, ln2_b, out);
        reset();
    }
}
